// VirtualNodeGNN_86870008529301
// MI455X (gfx1250) — matrix operand order repaired and checked by reading, not compiled, not run on hardware
//
#include <hip/hip_runtime.h>

// ---------------------------------------------------------------------------
// VirtualNodeGNN for MI455X (gfx1250, wave32, WMMA + TDM).
// Memory-bound overall (~2.5-3 GB/call -> ~120us floor @ 23.3 TB/s HBM).
// GEMMs: v_wmma_f32_16x16x32_bf16 (bf16 A/B post-LayerNorm, f32 accum).
// Weight tiles are DMA'd into LDS by the Tensor Data Mover (tensor_load_to_lds,
// TENSORcnt), with TDM LDS-padding generating the bank-conflict-avoiding
// 34-ushort row stride. Activations staged f32->bf16 via b128 loads.
// hn (102MB) fits the 192MB L2 -> edge gathers are L2 hits; softmax
// aggregation uses f32 atomics (avg degree ~2 => low contention).
// ---------------------------------------------------------------------------

#define Nn 100000
#define Ee 200000
#define Bb 2048
#define Hh 256

typedef __attribute__((ext_vector_type(16))) __bf16 v16bf;
typedef __attribute__((ext_vector_type(8)))  float  v8f;
typedef __attribute__((ext_vector_type(4)))  unsigned int v4u;
typedef __attribute__((ext_vector_type(8)))  int v8i_t;
typedef __attribute__((ext_vector_type(4)))  int v4i_t;

union Frag16 { unsigned int u[8]; v16bf v; };

__device__ __forceinline__ unsigned short f2bf(float x) {
  unsigned int u = __float_as_uint(x);
  u += 0x7FFFu + ((u >> 16) & 1u);   // round-to-nearest-even
  return (unsigned short)(u >> 16);
}

__device__ __forceinline__ void atomicMaxF(float* addr, float val) {
  if (val >= 0.0f) atomicMax((int*)addr, __float_as_int(val));
  else             atomicMin((unsigned int*)addr, __float_as_uint(val));
}

// ------------------------------ utility ------------------------------------

__global__ void fill_kernel(float* __restrict__ p, float v, int n) {
  int i = blockIdx.x * blockDim.x + threadIdx.x;
  if (i < n) p[i] = v;
}

// convert + transpose weights: in = 3 banks of [K][NC] f32, out = 3 banks of
// [NC][K] bf16 (so TDM tiles of Wt are plain rectangular copies).
template<int K, int NC>
__global__ __launch_bounds__(256) void convert_transpose_bf16_kernel(
    const float* __restrict__ in, unsigned short* __restrict__ out) {
  int idx = blockIdx.x * 256 + threadIdx.x;
  if (idx >= 3 * K * NC) return;
  int layer = idx / (K * NC);
  int rem = idx - layer * (K * NC);
  int k = rem / NC;
  int n = rem - k * NC;
  out[(size_t)layer * K * NC + (size_t)n * K + k] = f2bf(in[idx]);
}

__global__ void count_kernel(const int* __restrict__ batch, float* __restrict__ counts, int n) {
  int i = blockIdx.x * blockDim.x + threadIdx.x;
  if (i < n) atomicAdd(&counts[batch[i]], 1.0f);
}

__global__ __launch_bounds__(256) void vn_init_kernel(float* __restrict__ vn,
                                                      const float* __restrict__ vn_emb) {
  vn[(size_t)blockIdx.x * Hh + threadIdx.x] = vn_emb[threadIdx.x];
}

// ------------------------------ encoders -----------------------------------

template<int KIN>
__global__ __launch_bounds__(256) void encode_kernel(const float* __restrict__ xin,
                                                     const float* __restrict__ W,
                                                     const float* __restrict__ b,
                                                     float* __restrict__ outp) {
  int i = blockIdx.x, j = threadIdx.x;
  float acc = b[j];
#pragma unroll
  for (int k = 0; k < KIN; ++k)
    acc += xin[(size_t)i * KIN + k] * W[k * Hh + j];
  outp[(size_t)i * Hh + j] = acc;
}

// ------------------------- LayerNorm kernels -------------------------------

__global__ __launch_bounds__(256) void prenorm_relu_kernel(
    const float* __restrict__ h, const float* __restrict__ g, const float* __restrict__ b,
    float* __restrict__ hn, float* __restrict__ agg,
    float* __restrict__ segmax, float* __restrict__ denom) {
  __shared__ float red[256];
  int i = blockIdx.x, j = threadIdx.x;
  size_t off = (size_t)i * Hh + j;
  float v = h[off];
  red[j] = v; __syncthreads();
#pragma unroll
  for (int s = 128; s > 0; s >>= 1) { if (j < s) red[j] += red[j + s]; __syncthreads(); }
  float mu = red[0] * (1.0f / Hh);
  __syncthreads();
  float d = v - mu;
  red[j] = d * d; __syncthreads();
#pragma unroll
  for (int s = 128; s > 0; s >>= 1) { if (j < s) red[j] += red[j + s]; __syncthreads(); }
  float var = red[0] * (1.0f / Hh);
  float o = fmaxf(d * rsqrtf(var + 1e-5f) * g[j] + b[j], 0.0f);
  hn[off] = o;
  agg[off] = o;                      // root residual inside GENConv
  segmax[off] = -__builtin_inff();
  denom[off] = 0.0f;
}

__global__ __launch_bounds__(256) void mlp_ln_relu_kernel(
    float* __restrict__ y, const float* __restrict__ g, const float* __restrict__ b) {
  __shared__ float red[256];
  int i = blockIdx.x, j = threadIdx.x;
  size_t base = (size_t)i * 512;
  float v0 = y[base + j], v1 = y[base + j + 256];
  red[j] = v0 + v1; __syncthreads();
#pragma unroll
  for (int s = 128; s > 0; s >>= 1) { if (j < s) red[j] += red[j + s]; __syncthreads(); }
  float mu = red[0] * (1.0f / 512.0f);
  __syncthreads();
  float d0 = v0 - mu, d1 = v1 - mu;
  red[j] = d0 * d0 + d1 * d1; __syncthreads();
#pragma unroll
  for (int s = 128; s > 0; s >>= 1) { if (j < s) red[j] += red[j + s]; __syncthreads(); }
  float rs = rsqrtf(red[0] * (1.0f / 512.0f) + 1e-5f);
  y[base + j]       = fmaxf(d0 * rs * g[j] + b[j], 0.0f);
  y[base + j + 256] = fmaxf(d1 * rs * g[j + 256] + b[j + 256], 0.0f);
}

__global__ __launch_bounds__(256) void final_pool_kernel(
    const float* __restrict__ h, const float* __restrict__ g, const float* __restrict__ b,
    const int* __restrict__ batch, float* __restrict__ outp) {
  __shared__ float red[256];
  int i = blockIdx.x, j = threadIdx.x;
  size_t off = (size_t)i * Hh + j;
  float v = h[off];
  red[j] = v; __syncthreads();
#pragma unroll
  for (int s = 128; s > 0; s >>= 1) { if (j < s) red[j] += red[j + s]; __syncthreads(); }
  float mu = red[0] * (1.0f / Hh);
  __syncthreads();
  float d = v - mu;
  red[j] = d * d; __syncthreads();
#pragma unroll
  for (int s = 128; s > 0; s >>= 1) { if (j < s) red[j] += red[j + s]; __syncthreads(); }
  float var = red[0] * (1.0f / Hh);
  float o = fmaxf(d * rsqrtf(var + 1e-5f) * g[j] + b[j], 0.0f);
  atomicAdd(&outp[(size_t)batch[i] * Hh + j], o);
}

__global__ __launch_bounds__(256) void scale_out_kernel(float* __restrict__ outp,
                                                        const float* __restrict__ counts) {
  int bb = blockIdx.x, j = threadIdx.x;
  float inv = 1.0f / fmaxf(counts[bb], 1.0f);
  outp[(size_t)bb * Hh + j] *= inv;
}

// ------------------------ edge softmax aggregation -------------------------

__global__ __launch_bounds__(256) void edge_max_kernel(
    const float* __restrict__ hn, const float* __restrict__ ea,
    const int* __restrict__ src, const int* __restrict__ dst,
    const float* __restrict__ tptr, float* __restrict__ segmax) {
  int e = blockIdx.x, j = threadIdx.x;
  int s = src[e], d = dst[e];
  float m = fmaxf(hn[(size_t)s * Hh + j] + ea[(size_t)e * Hh + j], 0.0f) + 1e-7f;
  atomicMaxF(&segmax[(size_t)d * Hh + j], m * tptr[0]);
}

__global__ __launch_bounds__(256) void edge_expsum_kernel(
    const float* __restrict__ hn, const float* __restrict__ ea,
    const int* __restrict__ src, const int* __restrict__ dst,
    const float* __restrict__ tptr, const float* __restrict__ segmax,
    float* __restrict__ denom) {
  int e = blockIdx.x, j = threadIdx.x;
  int s = src[e], d = dst[e];
  size_t off = (size_t)d * Hh + j;
  float m = fmaxf(hn[(size_t)s * Hh + j] + ea[(size_t)e * Hh + j], 0.0f) + 1e-7f;
  float ex = __expf(m * tptr[0] - segmax[off]);
  atomicAdd(&denom[off], ex);
}

__global__ __launch_bounds__(256) void edge_aggregate_kernel(
    const float* __restrict__ hn, const float* __restrict__ ea,
    const int* __restrict__ src, const int* __restrict__ dst,
    const float* __restrict__ tptr, const float* __restrict__ segmax,
    const float* __restrict__ denom, float* __restrict__ agg) {
  int e = blockIdx.x, j = threadIdx.x;
  int s = src[e], d = dst[e];
  size_t off = (size_t)d * Hh + j;
  float m = fmaxf(hn[(size_t)s * Hh + j] + ea[(size_t)e * Hh + j], 0.0f) + 1e-7f;
  float ex = __expf(m * tptr[0] - segmax[off]);
  float alpha = ex / (denom[off] + 1e-16f);
  atomicAdd(&agg[off], m * alpha);
}

// ------------------------------ virtual node -------------------------------

__global__ __launch_bounds__(256) void vn_gather_kernel(
    const float* __restrict__ h, const int* __restrict__ batch, float* __restrict__ vn_acc) {
  int i = blockIdx.x, j = threadIdx.x;
  atomicAdd(&vn_acc[(size_t)batch[i] * Hh + j], h[(size_t)i * Hh + j]);
}

__global__ __launch_bounds__(256) void vn_update_kernel(
    float* __restrict__ vn, const float* __restrict__ vn_acc, const float* __restrict__ counts) {
  int bb = blockIdx.x, j = threadIdx.x;
  float inv = 1.0f / fmaxf(counts[bb], 1.0f);
  vn[(size_t)bb * Hh + j] += vn_acc[(size_t)bb * Hh + j] * inv;
}

__global__ __launch_bounds__(256) void add_vn_kernel(
    float* __restrict__ h, const int* __restrict__ batch, const float* __restrict__ vn) {
  int i = blockIdx.x, j = threadIdx.x;
  h[(size_t)i * Hh + j] += vn[(size_t)batch[i] * Hh + j];
}

// ------------------------------ WMMA GEMM ----------------------------------
// C[M,NC] = A[M,K] @ W[K,NC] + bias (ACCUM: +=), weights given pre-transposed
// and pre-converted: Wt[NC][K] bf16.
// 128x128 block tile, K-step 32, 8 waves = 4(M) x 2(N), per-wave 32x64 out.
// A: b128 global loads, f32->bf16 pack, ds_store_b32 into As[m][k] (pad 34).
// B: wave 0 issues a TDM tensor_load_to_lds of the 128x32 bf16 tile of Wt;
//    D# pad (1 DWORD every 16) produces the same 34-ushort row stride.

template<int K, int NC, bool ACCUM>
__global__ __launch_bounds__(256) void gemm_bf16_wmma(
    const float* __restrict__ A, const unsigned short* __restrict__ Wt,
    const float* __restrict__ bias, float* __restrict__ C, int M) {
  __shared__ unsigned short As[128][34];
  __shared__ unsigned short Bs[128][34];

  const int tid  = threadIdx.x;
  const int lane = tid & 31;
  const int wave = tid >> 5;
  const int wm   = wave & 3;     // 4 waves along M (32 rows each)
  const int wn   = wave >> 2;    // 2 waves along N (64 cols each)
  const int m0   = blockIdx.x * 128;
  const int n0   = blockIdx.y * 128;
  const int rsel = lane & 15;
  const int hsel = lane >> 4;
  const bool fullM = (m0 + 128 <= M);

  const v8f vzero = {0.f, 0.f, 0.f, 0.f, 0.f, 0.f, 0.f, 0.f};
  v8f acc[2][4];
#pragma unroll
  for (int mi = 0; mi < 2; ++mi)
#pragma unroll
    for (int ni = 0; ni < 4; ++ni) acc[mi][ni] = vzero;

  for (int kk = 0; kk < K; kk += 32) {
    // ---- stage A tile (128x32 f32 -> bf16), 4 floats per thread-iteration
    if (fullM) {
#pragma unroll
      for (int i = 0; i < 4; ++i) {
        int idx = tid + i * 256;          // 0..1023
        int r = idx >> 3;                 // 0..127
        int c = (idx & 7) * 4;            // 0,4,...,28
        const float4 a4 = *(const float4*)&A[(size_t)(m0 + r) * K + kk + c];
        unsigned int p0 = (unsigned int)f2bf(a4.x) | ((unsigned int)f2bf(a4.y) << 16);
        unsigned int p1 = (unsigned int)f2bf(a4.z) | ((unsigned int)f2bf(a4.w) << 16);
        *(unsigned int*)&As[r][c]     = p0;
        *(unsigned int*)&As[r][c + 2] = p1;
      }
    } else {
#pragma unroll
      for (int i = 0; i < 4; ++i) {
        int idx = tid + i * 256;
        int r = idx >> 3;
        int c = (idx & 7) * 4;
        float4 a4 = {0.f, 0.f, 0.f, 0.f};
        if (m0 + r < M) a4 = *(const float4*)&A[(size_t)(m0 + r) * K + kk + c];
        unsigned int p0 = (unsigned int)f2bf(a4.x) | ((unsigned int)f2bf(a4.y) << 16);
        unsigned int p1 = (unsigned int)f2bf(a4.z) | ((unsigned int)f2bf(a4.w) << 16);
        *(unsigned int*)&As[r][c]     = p0;
        *(unsigned int*)&As[r][c + 2] = p1;
      }
    }

    // ---- B tile via Tensor Data Mover: 128 rows x 32 bf16 from Wt[n0..][kk..]
    if (wave == 0) {
      unsigned long long ga =
          (unsigned long long)(const void*)(Wt + (size_t)n0 * K + kk);
      unsigned int lds = (unsigned int)(unsigned long long)(void*)&Bs[0][0];
      v4u g0;
      g0[0] = 1u;                                     // count=1 (valid user D#)
      g0[1] = lds;                                    // lds_addr
      g0[2] = (unsigned int)ga;                       // global_addr[31:0]
      g0[3] = (unsigned int)(ga >> 32) | (2u << 30);  // global_addr[56:32] | type=2
      v8i_t g1;
      g1[0] = (1 << 16)        // data_size = 2 bytes
            | (1 << 20)        // pad_enable
            | (3 << 22);       // pad_interval: every 16 DWORDs; pad_amount=0 -> 1 DWORD
      g1[1] = (int)(((unsigned)K & 0xFFFFu) << 16);               // tensor_dim0 lo
      g1[2] = (int)(((unsigned)K >> 16) | (((unsigned)NC & 0xFFFFu) << 16)); // dim0 hi | dim1 lo
      g1[3] = (int)(((unsigned)NC >> 16) | (32u << 16));          // dim1 hi | tile_dim0=32
      g1[4] = 128;                                                // tile_dim1=128 (tile_dim2=0)
      g1[5] = (int)K;                                             // tensor_dim0_stride lo
      g1[6] = 0;                                                  // stride hi | dim1_stride lo
      g1[7] = 0;
      v4i_t gz4 = {0, 0, 0, 0};                                   // dims >=2 unused
      v8i_t gz8 = {0, 0, 0, 0, 0, 0, 0, 0};
      // 6-arg form (this toolchain's clang): groups 0..3, extra group, cpol
      __builtin_amdgcn_tensor_load_to_lds(g0, g1, gz4, gz4, gz8, 0);
    }

    // prefetch next A tile (global_prefetch_b8)
    if (kk + 32 < K && fullM)
      __builtin_prefetch(&A[(size_t)(m0 + (tid >> 1)) * K + kk + 32], 0, 3);

    __builtin_amdgcn_s_wait_tensorcnt(0);   // no-op for non-issuing waves
    __syncthreads();

    // ---- fragments (ISA 16-bit layouts: lanes 0-15 K 0-7/16-23, 16-31 K 8-15/24-31)
    Frag16 af[2], bfr[4];
#pragma unroll
    for (int mi = 0; mi < 2; ++mi) {
      int r  = wm * 32 + mi * 16 + rsel;
      int kb = hsel * 8;
#pragma unroll
      for (int j = 0; j < 4; ++j)
        af[mi].u[j] = *(const unsigned int*)&As[r][kb + 2 * j];
#pragma unroll
      for (int j = 0; j < 4; ++j)
        af[mi].u[4 + j] = *(const unsigned int*)&As[r][16 + kb + 2 * j];
    }
#pragma unroll
    for (int ni = 0; ni < 4; ++ni) {
      int n  = wn * 64 + ni * 16 + rsel;
      int kb = hsel * 8;
#pragma unroll
      for (int j = 0; j < 8; ++j)
        bfr[ni].u[j] = *(const unsigned int*)&Bs[n][kb + 2 * j + ((j & 4) << 1)];
    }
#pragma unroll
    for (int mi = 0; mi < 2; ++mi)
#pragma unroll
      for (int ni = 0; ni < 4; ++ni)
        acc[mi][ni] = __builtin_amdgcn_wmma_f32_16x16x32_bf16(
            false, af[mi].v, false, bfr[ni].v, (short)0, acc[mi][ni], false, false);
    __syncthreads();
  }

  // ---- epilogue: C layout lane<16 -> M=vgpr, lane>=16 -> M=8+vgpr; N=lane&15
#pragma unroll
  for (int mi = 0; mi < 2; ++mi) {
#pragma unroll
    for (int ni = 0; ni < 4; ++ni) {
      int col = n0 + wn * 64 + ni * 16 + rsel;
      float bv = bias[col];
#pragma unroll
      for (int vj = 0; vj < 8; ++vj) {
        int row = m0 + wm * 32 + mi * 16 + vj + 8 * hsel;
        if (row < M) {
          float v = acc[mi][ni][vj] + bv;
          size_t off = (size_t)row * NC + col;
          if (ACCUM) C[off] += v; else C[off] = v;
        }
      }
    }
  }
}

// ------------------------------- driver ------------------------------------

extern "C" void kernel_launch(void* const* d_in, const int* in_sizes, int n_in,
                              void* d_out, int out_size, void* d_ws, size_t ws_size,
                              hipStream_t stream) {
  const float* x        = (const float*)d_in[0];
  const int*   eidx     = (const int*)d_in[1];
  const float* eattr    = (const float*)d_in[2];
  const int*   batch    = (const int*)d_in[3];
  const float* node_W   = (const float*)d_in[4];
  const float* node_b   = (const float*)d_in[5];
  const float* edge_W   = (const float*)d_in[6];
  const float* edge_b   = (const float*)d_in[7];
  const float* vn_emb   = (const float*)d_in[8];
  const float* ln_g     = (const float*)d_in[9];
  const float* ln_b     = (const float*)d_in[10];
  const float* tvals    = (const float*)d_in[11];
  const float* mlp_W1   = (const float*)d_in[12];
  const float* mlp_b1   = (const float*)d_in[13];
  const float* mlp_ln_g = (const float*)d_in[14];
  const float* mlp_ln_b = (const float*)d_in[15];
  const float* mlp_W2   = (const float*)d_in[16];
  const float* mlp_b2   = (const float*)d_in[17];
  float* out = (float*)d_out;

  const int* src = eidx;        // edge_index[0]
  const int* dst = eidx + Ee;   // edge_index[1]

  // workspace layout (f32 unless noted); segmax/denom overlay the GEMM1 y buffer
  const size_t NH = (size_t)Nn * Hh;
  float* h      = (float*)d_ws;
  float* hn     = h + NH;
  float* agg    = hn + NH;
  float* ybuf   = agg + NH;                 // N x 2H
  float* segmax = ybuf;
  float* denom  = ybuf + NH;
  float* ea     = ybuf + 2 * NH;            // E x H
  float* vn     = ea + (size_t)Ee * Hh;
  float* vn_acc = vn + (size_t)Bb * Hh;
  float* counts = vn_acc + (size_t)Bb * Hh;
  unsigned short* w1t = (unsigned short*)(counts + Bb);   // 3 x [512][256] bf16 (transposed)
  unsigned short* w2t = w1t + (size_t)3 * Hh * 2 * Hh;    // 3 x [256][512] bf16 (transposed)

  dim3 blk(256);
  const int WN = 3 * Hh * 2 * Hh;  // 393216 weights per MLP bank

  // weights -> bf16, transposed (recomputed every call: no cached state)
  convert_transpose_bf16_kernel<Hh, 2 * Hh><<<(WN + 255) / 256, blk, 0, stream>>>(mlp_W1, w1t);
  convert_transpose_bf16_kernel<2 * Hh, Hh><<<(WN + 255) / 256, blk, 0, stream>>>(mlp_W2, w2t);

  // batch counts and virtual-node init
  fill_kernel<<<(Bb + 255) / 256, blk, 0, stream>>>(counts, 0.0f, Bb);
  count_kernel<<<(Nn + 255) / 256, blk, 0, stream>>>(batch, counts, Nn);
  vn_init_kernel<<<Bb, blk, 0, stream>>>(vn, vn_emb);

  // encoders
  encode_kernel<9><<<Nn, blk, 0, stream>>>(x, node_W, node_b, h);
  encode_kernel<3><<<Ee, blk, 0, stream>>>(eattr, edge_W, edge_b, ea);

  for (int l = 1; l < 4; ++l) {
    prenorm_relu_kernel<<<Nn, blk, 0, stream>>>(h, ln_g + l * Hh, ln_b + l * Hh,
                                                hn, agg, segmax, denom);
    edge_max_kernel<<<Ee, blk, 0, stream>>>(hn, ea, src, dst, tvals + (l - 1), segmax);
    edge_expsum_kernel<<<Ee, blk, 0, stream>>>(hn, ea, src, dst, tvals + (l - 1),
                                               segmax, denom);
    edge_aggregate_kernel<<<Ee, blk, 0, stream>>>(hn, ea, src, dst, tvals + (l - 1),
                                                  segmax, denom, agg);

    dim3 g1((Nn + 127) / 128, (2 * Hh) / 128);
    gemm_bf16_wmma<Hh, 2 * Hh, false><<<g1, blk, 0, stream>>>(
        agg, w1t + (size_t)(l - 1) * Hh * 2 * Hh, mlp_b1 + (l - 1) * 2 * Hh, ybuf, Nn);
    mlp_ln_relu_kernel<<<Nn, blk, 0, stream>>>(ybuf, mlp_ln_g + (l - 1) * 2 * Hh,
                                               mlp_ln_b + (l - 1) * 2 * Hh);
    dim3 g2((Nn + 127) / 128, Hh / 128);
    gemm_bf16_wmma<2 * Hh, Hh, true><<<g2, blk, 0, stream>>>(
        ybuf, w2t + (size_t)(l - 1) * 2 * Hh * Hh, mlp_b2 + (l - 1) * Hh, h, Nn);

    // virtual node: scatter-mean of h, then broadcast back
    fill_kernel<<<(Bb * Hh + 255) / 256, blk, 0, stream>>>(vn_acc, 0.0f, Bb * Hh);
    vn_gather_kernel<<<Nn, blk, 0, stream>>>(h, batch, vn_acc);
    vn_update_kernel<<<Bb, blk, 0, stream>>>(vn, vn_acc, counts);
    add_vn_kernel<<<Nn, blk, 0, stream>>>(h, batch, vn);
  }

  // final norm + global mean pool
  fill_kernel<<<(Bb * Hh + 255) / 256, blk, 0, stream>>>(out, 0.0f, Bb * Hh);
  final_pool_kernel<<<Nn, blk, 0, stream>>>(h, ln_g, ln_b, batch, out);
  scale_out_kernel<<<Bb, blk, 0, stream>>>(out, counts);
}
